// RNNModel_11175504904559
// MI455X (gfx1250) — matrix operand order repaired and checked by reading, not compiled, not run on hardware
//
#include <hip/hip_runtime.h>
#include <hip/hip_bf16.h>

// Sizes
#define BB   32
#define LL   100
#define LP1  101
#define VV   50000
#define DD   256
#define HH   512
#define PP   8
#define G3H  1536           // 3*H
#define KIN  264            // D+P
#define KPAD 288            // padded to multiple of 32

typedef __attribute__((ext_vector_type(16))) __bf16 v16bf;
typedef __attribute__((ext_vector_type(8)))  float  v8f;

union V16U { uint4 q[2]; v16bf v; };

__device__ __forceinline__ v16bf load_a_chunk(const __bf16* __restrict__ arow,
                                              int kc, int lane) {
    // 16-bit A 16x32 layout: lanes 0-15 hold K=[0..7],[16..23]; lanes 16-31 K=[8..15],[24..31]
    int klo = (lane >> 4) * 8;
    V16U u;
    u.q[0] = *reinterpret_cast<const uint4*>(arow + kc + klo);
    u.q[1] = *reinterpret_cast<const uint4*>(arow + kc + klo + 16);
    return u.v;
}

__device__ __forceinline__ v16bf load_b_chunk(const __bf16* __restrict__ wrow,
                                              int kc, int lane) {
    // 16-bit B 32x16 layout: lanes 0-15 hold K=[0..7],[16..23]; lanes 16-31 K=[8..15],[24..31]
    int kb = (lane >> 4) * 8;
    V16U u;
    u.q[0] = *reinterpret_cast<const uint4*>(wrow + kc + kb);
    u.q[1] = *reinterpret_cast<const uint4*>(wrow + kc + kb + 16);
    return u.v;
}

// ---------------------------------------------------------------------------
// Branch-free strip body: 2 M-tiles x NV N-tiles (32 x NV*16 output block).
// D = A(bf16 MxK) * W(bf16 NxK)^T + bias, fp32 accumulate via WMMA.
// ---------------------------------------------------------------------------
template <int NV>
__device__ __forceinline__ void gemm_strip_body(const __bf16* __restrict__ A,
                                                const __bf16* __restrict__ W,
                                                const float*  __restrict__ bias,
                                                float* __restrict__ D,
                                                int N, int K, int m0, int nt0, int lane)
{
    int ncol = lane & 15;
    const __bf16* arow0 = A + (size_t)(m0      + ncol) * K;
    const __bf16* arow1 = A + (size_t)(m0 + 16 + ncol) * K;
    const __bf16* wrow[NV];
    #pragma unroll
    for (int t = 0; t < NV; ++t)
        wrow[t] = W + (size_t)((nt0 + t) * 16 + ncol) * K;

    const v8f vzero = {};
    v8f acc0[NV], acc1[NV];
    #pragma unroll
    for (int t = 0; t < NV; ++t) { acc0[t] = vzero; acc1[t] = vzero; }

    for (int kc = 0; kc < K; kc += 32) {
        v16bf a0 = load_a_chunk(arow0, kc, lane);
        v16bf a1 = load_a_chunk(arow1, kc, lane);
        #pragma unroll
        for (int t = 0; t < NV; ++t) {
            v16bf b = load_b_chunk(wrow[t], kc, lane);
            acc0[t] = __builtin_amdgcn_wmma_f32_16x16x32_bf16(false, a0, false, b, (short)0, acc0[t], false, false);
            acc1[t] = __builtin_amdgcn_wmma_f32_16x16x32_bf16(false, a1, false, b, (short)0, acc1[t], false, false);
        }
    }

    int rb0 = m0      + (lane >> 4) * 8;      // C/D layout: M = r + 8*(lane>=16)
    int rb1 = m0 + 16 + (lane >> 4) * 8;
    #pragma unroll
    for (int t = 0; t < NV; ++t) {
        int n = (nt0 + t) * 16 + ncol;
        float bv = bias[n];
        #pragma unroll
        for (int r = 0; r < 8; ++r) {
            D[(size_t)(rb0 + r) * N + n] = acc0[t][r] + bv;
            D[(size_t)(rb1 + r) * N + n] = acc1[t][r] + bv;
        }
    }
}

// One wave per 32 x 64 output block; scalar dispatch on N-remainder.
__global__ void gemm_bf16_strip(const __bf16* __restrict__ A,
                                const __bf16* __restrict__ W,
                                const float*  __restrict__ bias,
                                float* __restrict__ D,
                                int N, int K, int mstrips, int ntiles, int nstrips)
{
    int lane = threadIdx.x & 31;
    int wid  = __builtin_amdgcn_readfirstlane(
                   (int)(blockIdx.x * (blockDim.x >> 5) + (threadIdx.x >> 5)));
    int total = mstrips * nstrips;
    if (wid >= total) return;                 // scalar, wave-uniform exit
    int ms = wid / nstrips;
    int ns = wid - ms * nstrips;
    int m0  = ms * 32;
    int nt0 = ns * 4;
    int nvalid = ntiles - nt0; if (nvalid > 4) nvalid = 4;

    switch (nvalid) {
    case 4: gemm_strip_body<4>(A, W, bias, D, N, K, m0, nt0, lane); break;
    case 3: gemm_strip_body<3>(A, W, bias, D, N, K, m0, nt0, lane); break;
    case 2: gemm_strip_body<2>(A, W, bias, D, N, K, m0, nt0, lane); break;
    default: gemm_strip_body<1>(A, W, bias, D, N, K, m0, nt0, lane); break;
    }
}

// ---------------------------------------------------------------------------
// Fused GRU step: gh = h @ w_hh^T (3 gates via WMMA) + gate math.
// 64 waves: mt in {0,1} (batch tiles), nt in [0,32) (H tiles).
// ---------------------------------------------------------------------------
__global__ void gru_step(const __bf16* __restrict__ hb,    // h_cur bf16  [32x512]
                         const float*  __restrict__ hf,    // h_cur f32   [32x512]
                         const __bf16* __restrict__ whh,   // bf16 [1536x512]
                         const float*  __restrict__ gx,    // f32  [(L*B)x1536]
                         const float*  __restrict__ bhh,   // [1536]
                         float*  __restrict__ hf_n,
                         __bf16* __restrict__ hb_n,
                         __bf16* __restrict__ outs,        // bf16 [(B*L)x512]
                         int l)
{
    int lane = threadIdx.x & 31;
    int wid  = __builtin_amdgcn_readfirstlane(
                   (int)(blockIdx.x * (blockDim.x >> 5) + (threadIdx.x >> 5)));
    if (wid >= 64) return;
    int mt = wid & 1;
    int nt = wid >> 1;
    int m0 = mt * 16;
    int ncol = nt * 16 + (lane & 15);

    const __bf16* arow = hb + (size_t)(m0 + (lane & 15)) * HH;
    const __bf16* wr = whh + (size_t)(0 * HH * HH) + (size_t)ncol * HH;
    const __bf16* wz = whh + (size_t)(1 * HH * HH) + (size_t)ncol * HH;
    const __bf16* wn = whh + (size_t)(2 * HH * HH) + (size_t)ncol * HH;

    v8f accR = {}, accZ = {}, accN = {};
    #pragma unroll 4
    for (int kc = 0; kc < HH; kc += 32) {
        v16bf a  = load_a_chunk(arow, kc, lane);
        v16bf b0 = load_b_chunk(wr, kc, lane);
        accR = __builtin_amdgcn_wmma_f32_16x16x32_bf16(false, a, false, b0, (short)0, accR, false, false);
        v16bf b1 = load_b_chunk(wz, kc, lane);
        accZ = __builtin_amdgcn_wmma_f32_16x16x32_bf16(false, a, false, b1, (short)0, accZ, false, false);
        v16bf b2 = load_b_chunk(wn, kc, lane);
        accN = __builtin_amdgcn_wmma_f32_16x16x32_bf16(false, a, false, b2, (short)0, accN, false, false);
    }

    float bR = bhh[ncol], bZ = bhh[HH + ncol], bN = bhh[2 * HH + ncol];
    int rbase = m0 + (lane >> 4) * 8;
    #pragma unroll
    for (int r = 0; r < 8; ++r) {
        int b = rbase + r;
        const float* g = gx + ((size_t)l * BB + b) * G3H;
        float xr = g[ncol], xz = g[HH + ncol], xn = g[2 * HH + ncol];
        float hr = accR[r] + bR, hz = accZ[r] + bZ, hn = accN[r] + bN;
        float rg = 1.f / (1.f + __expf(-(xr + hr)));
        float z  = 1.f / (1.f + __expf(-(xz + hz)));
        float nn = tanhf(xn + rg * hn);
        float hp = hf[(size_t)b * HH + ncol];
        float h2 = (1.f - z) * nn + z * hp;
        hf_n[(size_t)b * HH + ncol] = h2;
        hb_n[(size_t)b * HH + ncol] = (__bf16)h2;
        outs[((size_t)b * LL + l) * HH + ncol] = (__bf16)h2;   // (B,L,H) row-major
    }
}

// ---------------------------------------------------------------------------
// Helpers: bf16 conversion with K padding, xin assembly, token mask scatter.
// ---------------------------------------------------------------------------
__global__ void cvt_pad_bf16(const float* __restrict__ src, __bf16* __restrict__ dst,
                             int rows, int Kin, int Kout)
{
    size_t i = (size_t)blockIdx.x * blockDim.x + threadIdx.x;
    size_t total = (size_t)rows * Kout;
    if (i >= total) return;
    int r = (int)(i / Kout), k = (int)(i - (size_t)r * Kout);
    dst[i] = (k < Kin) ? (__bf16)src[(size_t)r * Kin + k] : (__bf16)0.f;
}

__global__ void build_xin(const int* __restrict__ seq, const float* __restrict__ enc_w,
                          const float* __restrict__ pos_w, __bf16* __restrict__ xin)
{
    size_t i = (size_t)blockIdx.x * blockDim.x + threadIdx.x;
    if (i >= (size_t)LL * BB * KPAD) return;
    int row = (int)(i / KPAD), k = (int)(i - (size_t)row * KPAD);
    int l = row >> 5, b = row & 31;                  // row = l*32 + b
    float v;
    if (k < DD)            { int tok = seq[b * LP1 + l]; v = enc_w[(size_t)tok * DD + k]; }
    else if (k < KIN)      v = pos_w[l * PP + (k - DD)];
    else                   v = 0.f;
    xin[i] = (__bf16)v;
}

__global__ void mask_kernel(const int* __restrict__ seq, float* __restrict__ out)
{
    int b = blockIdx.x, l = blockIdx.y;
    float* row = out + ((size_t)b * LL + l) * VV;
    const float NEGINF = __int_as_float(0xff800000);
    for (int j = threadIdx.x; j < LL; j += blockDim.x) {
        int idx = (j <= l) ? seq[b * LP1 + j] : 0;   // j>l -> ids==0 per reference
        row[idx] = NEGINF;
    }
}

// ---------------------------------------------------------------------------
extern "C" void kernel_launch(void* const* d_in, const int* in_sizes, int n_in,
                              void* d_out, int out_size, void* d_ws, size_t ws_size,
                              hipStream_t stream)
{
    const int*   seq   = (const int*)  d_in[0];
    const float* enc_w = (const float*)d_in[1];
    const float* pos_w = (const float*)d_in[2];
    const float* w_ih  = (const float*)d_in[3];
    const float* w_hh  = (const float*)d_in[4];
    const float* b_ih  = (const float*)d_in[5];
    const float* b_hh  = (const float*)d_in[6];
    const float* dec_w = (const float*)d_in[7];
    const float* dec_b = (const float*)d_in[8];
    float* out = (float*)d_out;

    // workspace layout (256B aligned)
    char* ws = (char*)d_ws;
    size_t off = 0;
    auto take = [&](size_t bytes) { char* p = ws + off; off += (bytes + 255) & ~(size_t)255; return p; };
    __bf16* decw_bf = (__bf16*)take((size_t)VV * HH * 2);
    __bf16* xin_bf  = (__bf16*)take((size_t)LL * BB * KPAD * 2);
    __bf16* wih_bf  = (__bf16*)take((size_t)G3H * KPAD * 2);
    __bf16* whh_bf  = (__bf16*)take((size_t)G3H * HH * 2);
    float*  gx      = (float*) take((size_t)LL * BB * G3H * 4);
    __bf16* outs_bf = (__bf16*)take((size_t)BB * LL * HH * 2);
    float*  hf0     = (float*) take((size_t)BB * HH * 4);
    float*  hf1     = (float*) take((size_t)BB * HH * 4);
    __bf16* hb0     = (__bf16*)take((size_t)BB * HH * 2);
    __bf16* hb1     = (__bf16*)take((size_t)BB * HH * 2);
    (void)ws_size; (void)in_sizes; (void)n_in; (void)out_size;

    // 1) operand conversions to bf16 (+ zero padding)
    {
        size_t t = (size_t)VV * HH;
        cvt_pad_bf16<<<dim3((unsigned)((t + 255) / 256)), 256, 0, stream>>>(dec_w, decw_bf, VV, HH, HH);
    }
    {
        size_t t = (size_t)G3H * KPAD;
        cvt_pad_bf16<<<dim3((unsigned)((t + 255) / 256)), 256, 0, stream>>>(w_ih, wih_bf, G3H, KIN, KPAD);
    }
    {
        size_t t = (size_t)G3H * HH;
        cvt_pad_bf16<<<dim3((unsigned)((t + 255) / 256)), 256, 0, stream>>>(w_hh, whh_bf, G3H, HH, HH);
    }
    {
        size_t t = (size_t)LL * BB * KPAD;
        build_xin<<<dim3((unsigned)((t + 255) / 256)), 256, 0, stream>>>(seq, enc_w, pos_w, xin_bf);
    }
    // h0 = 0 (fp32 + bf16, buffers are contiguous: hf0,hf1,hb0,hb1)
    hipMemsetAsync(hf0, 0, (size_t)BB * HH * 4 * 2 + (size_t)BB * HH * 2 * 2, stream);

    // 2) gx = xin @ w_ih^T + b_ih : M=3200, N=1536, K=288
    {
        int mstrips = (LL * BB) / 32;           // 100
        int ntiles  = G3H / 16;                 // 96
        int nstrips = (ntiles + 3) / 4;         // 24
        int waves   = mstrips * nstrips;        // 2400
        gemm_bf16_strip<<<dim3((unsigned)((waves + 7) / 8)), 256, 0, stream>>>(
            xin_bf, wih_bf, b_ih, gx, G3H, KPAD, mstrips, ntiles, nstrips);
    }

    // 3) GRU scan: 100 dependent launches, ping-pong h buffers
    for (int l = 0; l < LL; ++l) {
        const __bf16* hb_c = (l & 1) ? hb1 : hb0;
        const float*  hf_c = (l & 1) ? hf1 : hf0;
        __bf16*       hb_n = (l & 1) ? hb0 : hb1;
        float*        hf_n = (l & 1) ? hf0 : hf1;
        gru_step<<<dim3(8), 256, 0, stream>>>(hb_c, hf_c, whh_bf, gx, b_hh,
                                              hf_n, hb_n, outs_bf, l);
    }

    // 4) decoder: result = outputs @ dec_w^T + dec_b : M=3200, N=50000, K=512
    {
        int mstrips = (BB * LL) / 32;           // 100
        int ntiles  = VV / 16;                  // 3125
        int nstrips = (ntiles + 3) / 4;         // 782
        int waves   = mstrips * nstrips;        // 78200
        gemm_bf16_strip<<<dim3((unsigned)((waves + 7) / 8)), 256, 0, stream>>>(
            outs_bf, decw_bf, dec_b, out, VV, HH, mstrips, ntiles, nstrips);
    }

    // 5) token mask (-inf scatter)
    mask_kernel<<<dim3(BB, LL), 128, 0, stream>>>(seq, out);

    // 6) hT (final h = hf0 after 100 steps) and enc_w passthrough
    float* out_hT  = out + (size_t)BB * LL * VV;
    float* out_enc = out_hT + (size_t)BB * HH;
    hipMemcpyAsync(out_hT,  hf0,  (size_t)BB * HH * 4,  hipMemcpyDeviceToDevice, stream);
    hipMemcpyAsync(out_enc, enc_w, (size_t)VV * DD * 4, hipMemcpyDeviceToDevice, stream);
}
